// MotherNetRegression_89498528514765
// MI455X (gfx1250) — matrix operand order repaired and checked by reading, not compiled, not run on hardware
//
#include <hip/hip_runtime.h>

#define T_ROWS 4096
#define NB 8
#define NF 512
#define NH 1024

typedef __attribute__((ext_vector_type(16))) __bf16 v16bf;
typedef __attribute__((ext_vector_type(8)))  float  v8f;
typedef unsigned short u16;

union V16U { v16bf v; uint4 q[2]; u16 u[16]; };

// round-to-nearest-even f32 -> bf16 (raw u16)
__device__ __forceinline__ u16 f2bf_rn(float f) {
    unsigned int u = __float_as_uint(f);
    u += 0x7fffu + ((u >> 16) & 1u);
    return (u16)(u >> 16);
}
__device__ __forceinline__ float bf2f(u16 h) {
    return __uint_as_float(((unsigned int)h) << 16);
}
__device__ __forceinline__ void split2(float f, u16& hi, u16& lo) {
    u16 h = f2bf_rn(f);
    hi = h;
    lo = f2bf_rn(f - bf2f(h));
}

// ---------------------------------------------------------------------------
// Prep 1: x [T,B,F] f32 -> xhi/xlo bf16 (with nan_to_num). 4 elems/thread.
// ---------------------------------------------------------------------------
__global__ __launch_bounds__(256) void prep_x(
    const float* __restrict__ x, u16* __restrict__ xhi, u16* __restrict__ xlo)
{
    const int i = blockIdx.x * 256 + threadIdx.x;     // i indexes float4 groups
    const float4 f4 = reinterpret_cast<const float4*>(x)[i];
    float f[4] = {f4.x, f4.y, f4.z, f4.w};
    u16 h[4], l[4];
#pragma unroll
    for (int j = 0; j < 4; ++j) {
        float v = f[j];
        v = (v == v) ? v : 0.0f;                      // jnp.nan_to_num
        split2(v, h[j], l[j]);
    }
    uint2 ph, pl;
    ph.x = (unsigned)h[0] | ((unsigned)h[1] << 16);
    ph.y = (unsigned)h[2] | ((unsigned)h[3] << 16);
    pl.x = (unsigned)l[0] | ((unsigned)l[1] << 16);
    pl.y = (unsigned)l[2] | ((unsigned)l[3] << 16);
    reinterpret_cast<uint2*>(xhi)[i] = ph;
    reinterpret_cast<uint2*>(xlo)[i] = pl;
}

// ---------------------------------------------------------------------------
// Prep 2: w1 [B,F,H] f32 -> w1t_hi/lo [B,H,F] bf16 (LDS-tiled transpose).
// grid (H/32, F/32, B), block (32, 8).
// ---------------------------------------------------------------------------
__global__ __launch_bounds__(256) void prep_w1(
    const float* __restrict__ w1, u16* __restrict__ whi, u16* __restrict__ wlo)
{
    __shared__ float tile[32][33];
    const int b  = blockIdx.z;
    const int h0 = blockIdx.x * 32;
    const int f0 = blockIdx.y * 32;
    const int tx = threadIdx.x, ty = threadIdx.y;

#pragma unroll
    for (int r = ty; r < 32; r += 8)                   // coalesced over h
        tile[r][tx] = w1[((size_t)b * NF + f0 + r) * NH + h0 + tx];
    __syncthreads();

#pragma unroll
    for (int r = ty; r < 32; r += 8) {                 // coalesced over f
        float v = tile[tx][r];                         // = w1[b][f0+tx][h0+r]
        u16 hi, lo;
        split2(v, hi, lo);
        const size_t o = ((size_t)b * NH + h0 + r) * NF + f0 + tx;
        whi[o] = hi;
        wlo[o] = lo;
    }
}

// ---------------------------------------------------------------------------
// Main fused kernel (pre-split operands): inner loop = 8x b128 load + 3x WMMA
// ---------------------------------------------------------------------------
__global__ __launch_bounds__(128) void mothernet_wmma_main(
    const u16* __restrict__ xhi, const u16* __restrict__ xlo,   // [T,B,F] bf16
    const u16* __restrict__ whi, const u16* __restrict__ wlo,   // [B,H,F] bf16
    const float* __restrict__ b1, const float* __restrict__ w2,
    const float* __restrict__ b2, float* __restrict__ out)
{
    const int b    = blockIdx.y;
    const int lane = threadIdx.x & 31;
    const int wave = threadIdx.x >> 5;                 // 4 waves / block
    const int grp  = lane >> 4;                        // half-wave id
    const int l16  = lane & 15;
    const int m_base = blockIdx.x * 64 + wave * 16;

    // A fragment: row M = lane%16; K runs {c..c+7, c+16..c+23}, c = grp*8
    const size_t rowoff = ((size_t)(m_base + l16) * NB + b) * NF + grp * 8;
    const u16* xh_row = xhi + rowoff;
    const u16* xl_row = xlo + rowoff;

    float outacc[8];
#pragma unroll
    for (int r = 0; r < 8; ++r) outacc[r] = 0.0f;

#pragma unroll 1
    for (int n0 = 0; n0 < NH; n0 += 16) {
        const int hcol = n0 + l16;                     // B column N = lane%16
        v8f acc = {};

        // B fragment: column h, K = kb + grp*8 + {0..7, 16..23} in w1t
        const size_t coloff = ((size_t)b * NH + hcol) * NF + grp * 8;
        const u16* bh_col = whi + coloff;
        const u16* bl_col = wlo + coloff;

#pragma unroll 2
        for (int kb = 0; kb < NF; kb += 32) {
            V16U ah, al, bh, bl;
            ah.q[0] = *reinterpret_cast<const uint4*>(xh_row + kb);
            ah.q[1] = *reinterpret_cast<const uint4*>(xh_row + kb + 16);
            al.q[0] = *reinterpret_cast<const uint4*>(xl_row + kb);
            al.q[1] = *reinterpret_cast<const uint4*>(xl_row + kb + 16);
            bh.q[0] = *reinterpret_cast<const uint4*>(bh_col + kb);
            bh.q[1] = *reinterpret_cast<const uint4*>(bh_col + kb + 16);
            bl.q[0] = *reinterpret_cast<const uint4*>(bl_col + kb);
            bl.q[1] = *reinterpret_cast<const uint4*>(bl_col + kb + 16);

            // bf16x3: x*w ~= xl*wh + xh*wl + xh*wh (f32 accumulate)
            acc = __builtin_amdgcn_wmma_f32_16x16x32_bf16(
                      false, al.v, false, bh.v, (short)0, acc, false, false);
            acc = __builtin_amdgcn_wmma_f32_16x16x32_bf16(
                      false, ah.v, false, bl.v, (short)0, acc, false, false);
            acc = __builtin_amdgcn_wmma_f32_16x16x32_bf16(
                      false, ah.v, false, bh.v, (short)0, acc, false, false);
        }

        const float b1v = b1[b * NH + hcol];
        const float w2v = w2[b * NH + hcol];
#pragma unroll
        for (int r = 0; r < 8; ++r) {
            float hv = acc[r] + b1v;
            hv = hv > 0.0f ? hv : 0.0f;                // relu
            outacc[r] = fmaf(hv, w2v, outacc[r]);      // layer 2 (O==1)
        }
    }

#pragma unroll
    for (int r = 0; r < 8; ++r) {
        float v = outacc[r];
        v += __shfl_xor(v, 1, 32);
        v += __shfl_xor(v, 2, 32);
        v += __shfl_xor(v, 4, 32);
        v += __shfl_xor(v, 8, 32);
        outacc[r] = v;
    }

    if (l16 == 0) {
        const float b2v = b2[b];
#pragma unroll
        for (int r = 0; r < 8; ++r) {
            const int t = m_base + grp * 8 + r;        // M = r (+8 for hi half)
            out[(size_t)t * NB + b] = outacc[r] + b2v;
        }
    }
}

// ---------------------------------------------------------------------------
// Fallback (no workspace): convert inside the loop. Same math, slower.
// ---------------------------------------------------------------------------
__global__ __launch_bounds__(128) void mothernet_wmma_fallback(
    const float* __restrict__ x, const float* __restrict__ w1,
    const float* __restrict__ b1, const float* __restrict__ w2,
    const float* __restrict__ b2, float* __restrict__ out)
{
    const int b    = blockIdx.y;
    const int lane = threadIdx.x & 31;
    const int wave = threadIdx.x >> 5;
    const int grp  = lane >> 4;
    const int l16  = lane & 15;
    const int m_base = blockIdx.x * 64 + wave * 16;

    const float* xrow = x + ((size_t)(m_base + l16) * NB + b) * NF;

    float outacc[8];
#pragma unroll
    for (int r = 0; r < 8; ++r) outacc[r] = 0.0f;

#pragma unroll 1
    for (int n0 = 0; n0 < NH; n0 += 16) {
        const int hcol = n0 + l16;
        v8f acc = {};
        const float* wcol = w1 + ((size_t)b * NF + (size_t)grp * 8) * NH + hcol;

#pragma unroll 1
        for (int kb = 0; kb < NF; kb += 32) {
            const float4* p = reinterpret_cast<const float4*>(xrow + kb + grp * 8);
            const float4* q = reinterpret_cast<const float4*>(xrow + kb + 16 + grp * 8);
            float4 t0 = p[0], t1 = p[1], t2 = q[0], t3 = q[1];
            float af[16] = {t0.x,t0.y,t0.z,t0.w, t1.x,t1.y,t1.z,t1.w,
                            t2.x,t2.y,t2.z,t2.w, t3.x,t3.y,t3.z,t3.w};
            V16U ahv, alv, bhv, blv;
#pragma unroll
            for (int j = 0; j < 16; ++j) {
                float f = af[j];
                f = (f == f) ? f : 0.0f;
                split2(f, ahv.u[j], alv.u[j]);
            }
#pragma unroll
            for (int j = 0; j < 16; ++j) {
                float f = wcol[(size_t)(kb + j + (j & 8)) * NH];
                split2(f, bhv.u[j], blv.u[j]);
            }
            acc = __builtin_amdgcn_wmma_f32_16x16x32_bf16(
                      false, alv.v, false, bhv.v, (short)0, acc, false, false);
            acc = __builtin_amdgcn_wmma_f32_16x16x32_bf16(
                      false, ahv.v, false, blv.v, (short)0, acc, false, false);
            acc = __builtin_amdgcn_wmma_f32_16x16x32_bf16(
                      false, ahv.v, false, bhv.v, (short)0, acc, false, false);
        }

        const float b1v = b1[b * NH + hcol];
        const float w2v = w2[b * NH + hcol];
#pragma unroll
        for (int r = 0; r < 8; ++r) {
            float hv = acc[r] + b1v;
            hv = hv > 0.0f ? hv : 0.0f;
            outacc[r] = fmaf(hv, w2v, outacc[r]);
        }
    }

#pragma unroll
    for (int r = 0; r < 8; ++r) {
        float v = outacc[r];
        v += __shfl_xor(v, 1, 32);
        v += __shfl_xor(v, 2, 32);
        v += __shfl_xor(v, 4, 32);
        v += __shfl_xor(v, 8, 32);
        outacc[r] = v;
    }

    if (l16 == 0) {
        const float b2v = b2[b];
#pragma unroll
        for (int r = 0; r < 8; ++r) {
            const int t = m_base + grp * 8 + r;
            out[(size_t)t * NB + b] = outacc[r] + b2v;
        }
    }
}

extern "C" void kernel_launch(void* const* d_in, const int* in_sizes, int n_in,
                              void* d_out, int out_size, void* d_ws, size_t ws_size,
                              hipStream_t stream) {
    (void)in_sizes; (void)n_in; (void)out_size;
    const float* x  = (const float*)d_in[0];  // [4096, 8, 512]
    const float* w1 = (const float*)d_in[1];  // [8, 512, 1024]
    const float* b1 = (const float*)d_in[2];  // [8, 1024]
    const float* w2 = (const float*)d_in[3];  // [8, 1024, 1]
    const float* b2 = (const float*)d_in[4];  // [8, 1]
    float* out = (float*)d_out;               // [4096, 8, 1]

    const size_t NX = (size_t)T_ROWS * NB * NF;       // 16,777,216
    const size_t NW = (size_t)NB * NF * NH;           //  4,194,304
    const size_t WS_NEEDED = (NX * 2 + NW * 2) * sizeof(u16);  // 80 MB

    if (ws_size >= WS_NEEDED) {
        u16* xhi = (u16*)d_ws;
        u16* xlo = xhi + NX;
        u16* whi = xlo + NX;
        u16* wlo = whi + NW;

        prep_x<<<dim3((unsigned)(NX / 4 / 256)), dim3(256), 0, stream>>>(x, xhi, xlo);
        prep_w1<<<dim3(NH / 32, NF / 32, NB), dim3(32, 8), 0, stream>>>(w1, whi, wlo);

        dim3 grid(T_ROWS / 64, NB);                   // 64 x 8 blocks
        dim3 block(128);                              // 4 waves (wave32)
        mothernet_wmma_main<<<grid, block, 0, stream>>>(
            xhi, xlo, whi, wlo, b1, w2, b2, out);
    } else {
        dim3 grid(T_ROWS / 64, NB);
        dim3 block(128);
        mothernet_wmma_fallback<<<grid, block, 0, stream>>>(x, w1, b1, w2, b2, out);
    }
}
